// TransformerEncoder_24386824306896
// MI455X (gfx1250) — matrix operand order repaired and checked by reading, not compiled, not run on hardware
//
#include <hip/hip_runtime.h>
#include <math.h>

#define B_   32
#define S_   512
#define T_   513
#define D_   512
#define H_   8
#define DH_  64
#define FF_  2048
#define L_   4
#define MT_   (T_ * B_)   // 16416 transformer rows
#define MTOK_ (S_ * B_)   // 16384 token rows

typedef __attribute__((ext_vector_type(16))) __bf16       v16bf;
typedef __attribute__((ext_vector_type(8)))  float        v8f;
typedef __attribute__((ext_vector_type(4)))  unsigned int u32x4;
typedef __attribute__((ext_vector_type(8)))  int          i32x8;
typedef __attribute__((ext_vector_type(4)))  int          i32x4;

union Frag { v16bf v; u32x4 q[2]; };

__device__ __forceinline__ unsigned short f2bf(float f) {
  unsigned int u = __float_as_uint(f);
  u += 0x7FFFu + ((u >> 16) & 1u);           // round-to-nearest-even
  return (unsigned short)(u >> 16);
}

__device__ __forceinline__ v8f vzero8() {
  v8f z;
#pragma unroll
  for (int i = 0; i < 8; ++i) z[i] = 0.f;
  return z;
}

__device__ __forceinline__ v8f wmma_bf16(const Frag& a, const Frag& b, v8f c) {
  return __builtin_amdgcn_wmma_f32_16x16x32_bf16(false, a.v, false, b.v,
                                                 (short)0, c, false, false);
}

// ---------------------------------------------------------------------------
// TDM: load a 64x32 bf16 tile (rows x k) from a row-major matrix with row
// stride K elements into LDS, padding each 16-DWORD row with 4 DWORDs so the
// LDS layout is exactly sA[64][40] (stride 40 halfwords = 80 bytes).
// D# bit layout per CDNA5 ISA 8.3-8.6. rows_valid rows OOB-checked (zero fill).
// Uses the 6-arg builtin form (clang-23 / therock headers).
// ---------------------------------------------------------------------------
__device__ __forceinline__ void tdm_load_tile_a(const unsigned short* gaddr,
                                                unsigned int lds_off,
                                                unsigned int rows_valid,
                                                unsigned int K) {
  unsigned long long ga = (unsigned long long)(uintptr_t)gaddr;
  u32x4 g0;
  g0[0] = 1u;                                              // count=1 (valid), user mode
  g0[1] = lds_off;                                         // lds_addr (bytes)
  g0[2] = (unsigned int)(ga & 0xFFFFFFFFu);                // global_addr[31:0]
  g0[3] = (unsigned int)((ga >> 32) & 0x01FFFFFFu)         // global_addr[56:32]
        | (2u << 30);                                      // type = 2 ("image")
  i32x8 g1;
  g1[0] = (int)((1u << 16)      // data_size = 2 bytes (bf16)
              | (1u << 20)      // pad_enable
              | (3u << 22)      // pad_interval: 2^(3+1)=16 DWORDs (one 32-elem row)
              | (3u << 25));    // pad_amount: 3+1=4 DWORDs (16B) -> stride 40 halfwords
  unsigned int td0 = K;                                    // tensor_dim0 (elements)
  unsigned int td1 = rows_valid;                           // tensor_dim1 (rows; OOB -> 0)
  g1[1] = (int)((td0 & 0xFFFFu) << 16);                    // [47:32]=atomic_barrier=0, [63:48]=td0.lo
  g1[2] = (int)((td0 >> 16) | ((td1 & 0xFFFFu) << 16));    // [79:64]=td0.hi, [95:80]=td1.lo
  g1[3] = (int)((td1 >> 16) | (32u << 16));                // [111:96]=td1.hi, [127:112]=tile_dim0=32
  g1[4] = (int)64u;                                        // [143:128]=tile_dim1=64, tile_dim2=0
  g1[5] = (int)K;                                          // tensor_dim0_stride[31:0] (elements)
  g1[6] = 0;                                               // stride[47:32]=0, dim1_stride.lo=0
  g1[7] = 0;
  i32x4 z4; z4[0] = z4[1] = z4[2] = z4[3] = 0;             // groups 2/3: 2D tile
  i32x8 z8;
#pragma unroll
  for (int i = 0; i < 8; ++i) z8[i] = 0;
  __builtin_amdgcn_tensor_load_to_lds(g0, g1, z4, z4, z8, 0);
}

// ---------------------------------------------------------------------------
// Generic bf16 WMMA GEMM: C[M,N] = epilogue(A[M,K] @ W[K,N] + bias [+ resid])
// Block tile 64x64, BK=32, 128 threads = 4 waves, each wave 32x32 (2x2 WMMA).
// A tile staged by the Tensor Data Mover (overlapped with the vector-path
// transpose staging of the B tile). ACT: 0 = none, 1 = exact GELU.
// ---------------------------------------------------------------------------
template<int ACT, bool RESID, bool O32, bool O16>
__global__ __launch_bounds__(128)
void gemm_bf16_k(const unsigned short* __restrict__ A,
                 const unsigned short* __restrict__ W,
                 const float* __restrict__ bias,
                 const float* __restrict__ resid,
                 float* __restrict__ C32,
                 unsigned short* __restrict__ C16,
                 int M, int N, int K) {
  __shared__ unsigned short sA[64][40];  // rows x k   (stride 40 -> 16B aligned, bank-spread)
  __shared__ unsigned short sB[64][40];  // cols(n) x k

  const int tid  = threadIdx.x;
  const int lane = tid & 31;
  const int wave = tid >> 5;
  const int wm   = (wave & 1) * 32;
  const int wn   = (wave >> 1) * 32;
  const int half = lane >> 4;
  const int l15  = lane & 15;
  const int m0   = blockIdx.y * 64;
  const int n0   = blockIdx.x * 64;
  const unsigned int ldsA = (unsigned int)(uintptr_t)&sA[0][0];  // LDS byte offset

  v8f acc[2][2];
#pragma unroll
  for (int i = 0; i < 2; ++i)
#pragma unroll
    for (int j = 0; j < 2; ++j) acc[i][j] = vzero8();

  for (int k0 = 0; k0 < K; k0 += 32) {
    // ---- TDM: async DMA of the A tile (64x32) into padded LDS ----
    if (wave == 0) {
      tdm_load_tile_a(A + (size_t)m0 * K + k0, ldsA,
                      (unsigned int)(M - m0), (unsigned int)K);
    }
    // ---- vector path: stage B (32x64) transposed to n-major (overlaps TDM) ----
#pragma unroll
    for (int it = 0; it < 2; ++it) {
      int c = it * 128 + tid;
      int bk = c >> 3;
      int bn = (c & 7) * 8;
      u32x4 bv = *(const u32x4*)(W + (size_t)(k0 + bk) * N + n0 + bn);
#pragma unroll
      for (int j = 0; j < 4; ++j) {
        unsigned int dw = bv[j];
        sB[bn + 2 * j][bk]     = (unsigned short)(dw & 0xFFFFu);
        sB[bn + 2 * j + 1][bk] = (unsigned short)(dw >> 16);
      }
    }
    if (k0 + 32 < K) {  // prefetch next W tile (global_prefetch_b8)
      __builtin_prefetch(W + (size_t)(k0 + 32 + (tid & 31)) * N + n0, 0, 1);
    }
    if (wave == 0) __builtin_amdgcn_s_wait_tensorcnt(0);
    __syncthreads();

    // ---- fragments + WMMA ----
    Frag a[2], bf[2];
#pragma unroll
    for (int i = 0; i < 2; ++i) {
      const unsigned short* p = &sA[wm + i * 16 + l15][half * 8];
      a[i].q[0] = *(const u32x4*)p;        // K = half*8 + 0..7
      a[i].q[1] = *(const u32x4*)(p + 16); // K = 16 + half*8 + 0..7
    }
#pragma unroll
    for (int j = 0; j < 2; ++j) {
      const unsigned short* p = &sB[wn + j * 16 + l15][half * 8];
      bf[j].q[0] = *(const u32x4*)p;       // K = half*8 + 0..7
      bf[j].q[1] = *(const u32x4*)(p + 16); // K = 16 + half*8 + 0..7
    }
#pragma unroll
    for (int i = 0; i < 2; ++i)
#pragma unroll
      for (int j = 0; j < 2; ++j) acc[i][j] = wmma_bf16(a[i], bf[j], acc[i][j]);
    __syncthreads();
  }

  // ---- epilogue ----
#pragma unroll
  for (int i = 0; i < 2; ++i)
#pragma unroll
    for (int j = 0; j < 2; ++j) {
      int col = n0 + wn + j * 16 + l15;
      float bv = bias[col];
#pragma unroll
      for (int g = 0; g < 8; ++g) {
        int row = m0 + wm + i * 16 + half * 8 + g;
        if (row < M) {
          float v = acc[i][j][g] + bv;
          if (RESID) v += resid[(size_t)row * N + col];
          if (ACT == 1) v = 0.5f * v * (1.f + erff(v * 0.70710678118f));
          if (O32) C32[(size_t)row * N + col] = v;
          if (O16) C16[(size_t)row * N + col] = f2bf(v);
        }
      }
    }
}

// ---------------------------------------------------------------------------
// Flash attention: one (b,h) per block, 64 query rows, 4 waves x 16 rows,
// streaming 32-key blocks. qkv layout: [t*B+b][3*D], q|k|v segments.
// ---------------------------------------------------------------------------
__global__ __launch_bounds__(128)
void attn_kernel(const unsigned short* __restrict__ qkv,
                 unsigned short* __restrict__ o) {
  __shared__ unsigned short sK[32][72];      // keys x d (row-major)
  __shared__ unsigned short sV[64][40];      // d x keys (transposed)
  __shared__ unsigned short sP[4][16][40];   // per-wave P staging

  const int tid  = threadIdx.x;
  const int w    = tid >> 5;
  const int lane = tid & 31;
  const int half = lane >> 4;
  const int l15  = lane & 15;
  const int b    = blockIdx.x >> 3;
  const int h    = blockIdx.x & 7;
  const int tq0  = blockIdx.y * 64;

  // Q fragments, resident in registers (two 16x32 A-frags covering d 0..63)
  int qt = tq0 + w * 16 + l15; if (qt > T_ - 1) qt = T_ - 1;
  const unsigned short* qp = qkv + ((size_t)qt * B_ + b) * (3 * D_) + h * DH_;
  Frag aq[2];
  aq[0].q[0] = *(const u32x4*)(qp + half * 8);
  aq[0].q[1] = *(const u32x4*)(qp + 16 + half * 8);
  aq[1].q[0] = *(const u32x4*)(qp + 32 + half * 8);
  aq[1].q[1] = *(const u32x4*)(qp + 48 + half * 8);

  v8f oacc[4];
#pragma unroll
  for (int n = 0; n < 4; ++n) oacc[n] = vzero8();
  float mi[8], li[8];
#pragma unroll
  for (int g = 0; g < 8; ++g) { mi[g] = -3e30f; li[g] = 0.f; }
  const float scale = 0.125f;  // 1/sqrt(64)

  for (int kb = 0; kb < 17; ++kb) {  // 17*32 = 544 >= 513 keys
    // ---- stage K tile and transposed V tile ----
#pragma unroll
    for (int it = 0; it < 2; ++it) {
      int c = it * 128 + tid;
      int kr = c >> 3;
      int doff = (c & 7) * 8;
      int t = kb * 32 + kr;
      u32x4 kv, vv;
      if (t < T_) {
        const unsigned short* base = qkv + ((size_t)t * B_ + b) * (3 * D_) + h * DH_ + doff;
        kv = *(const u32x4*)(base + D_);
        vv = *(const u32x4*)(base + 2 * D_);
      } else {
        kv[0] = kv[1] = kv[2] = kv[3] = 0u;
        vv[0] = vv[1] = vv[2] = vv[3] = 0u;
      }
      *(u32x4*)&sK[kr][doff] = kv;
#pragma unroll
      for (int j = 0; j < 4; ++j) {
        unsigned int dw = vv[j];
        sV[doff + 2 * j][kr]     = (unsigned short)(dw & 0xFFFFu);
        sV[doff + 2 * j + 1][kr] = (unsigned short)(dw >> 16);
      }
    }
    __syncthreads();

    // ---- S = Q @ K^T : 16 x 32 logits per wave ----
    v8f s0 = vzero8(), s1 = vzero8();
#pragma unroll
    for (int kk = 0; kk < 2; ++kk) {
      Frag bk0, bk1;
      const unsigned short* p0 = &sK[l15][kk * 32 + half * 8];
      bk0.q[0] = *(const u32x4*)p0; bk0.q[1] = *(const u32x4*)(p0 + 16);
      const unsigned short* p1 = &sK[16 + l15][kk * 32 + half * 8];
      bk1.q[0] = *(const u32x4*)p1; bk1.q[1] = *(const u32x4*)(p1 + 16);
      s0 = wmma_bf16(aq[kk], bk0, s0);
      s1 = wmma_bf16(aq[kk], bk1, s1);
    }

    // ---- online softmax update ----
    bool ok0 = (kb * 32 + l15)      < T_;
    bool ok1 = (kb * 32 + 16 + l15) < T_;
#pragma unroll
    for (int g = 0; g < 8; ++g) {
      float v0 = ok0 ? s0[g] * scale : -3e30f;
      float v1 = ok1 ? s1[g] * scale : -3e30f;
      float rm = fmaxf(v0, v1);
#pragma unroll
      for (int msk = 1; msk < 16; msk <<= 1) rm = fmaxf(rm, __shfl_xor(rm, msk, 32));
      float mnew  = fmaxf(mi[g], rm);
      float alpha = __expf(mi[g] - mnew);
      float p0    = __expf(v0 - mnew);
      float p1    = __expf(v1 - mnew);
      float rs    = p0 + p1;
#pragma unroll
      for (int msk = 1; msk < 16; msk <<= 1) rs += __shfl_xor(rs, msk, 32);
      mi[g] = mnew;
      li[g] = li[g] * alpha + rs;
#pragma unroll
      for (int n = 0; n < 4; ++n) oacc[n][g] *= alpha;
      int rloc = g + 8 * half;   // C-layout row for this lane
      sP[w][rloc][l15]      = f2bf(p0);
      sP[w][rloc][16 + l15] = f2bf(p1);
    }

    // ---- O += P @ V (P reloaded in A-frag layout; wave-local LDS) ----
    Frag ap;
    const unsigned short* pp = &sP[w][l15][half * 8];
    ap.q[0] = *(const u32x4*)pp;
    ap.q[1] = *(const u32x4*)(pp + 16);
#pragma unroll
    for (int n = 0; n < 4; ++n) {
      Frag bv;
      const unsigned short* pv = &sV[n * 16 + l15][half * 8];
      bv.q[0] = *(const u32x4*)pv; bv.q[1] = *(const u32x4*)(pv + 16);
      oacc[n] = wmma_bf16(ap, bv, oacc[n]);
    }
    __syncthreads();
  }

  // ---- normalize + store bf16 o[t*B+b][h*64+d] ----
#pragma unroll
  for (int n = 0; n < 4; ++n)
#pragma unroll
    for (int g = 0; g < 8; ++g) {
      int t = tq0 + w * 16 + g + 8 * half;
      if (t < T_) {
        float v = oacc[n][g] / li[g];
        o[((size_t)t * B_ + b) * D_ + h * DH_ + n * 16 + l15] = f2bf(v);
      }
    }
}

// ---------------------------------------------------------------------------
// LayerNorm over D=512 (optionally fused leaky-relu), writes f32 and/or bf16.
// ---------------------------------------------------------------------------
template<bool LEAKY>
__global__ __launch_bounds__(128)
void ln_kernel(const float* __restrict__ X, const float* __restrict__ g,
               const float* __restrict__ bb, float* __restrict__ o32,
               unsigned short* __restrict__ o16, int ld16) {
  __shared__ float red[8];
  const int r = blockIdx.x, tid = threadIdx.x;
  const float* xr = X + (size_t)r * D_;
  float x[4], s1 = 0.f, s2 = 0.f;
#pragma unroll
  for (int i = 0; i < 4; ++i) { x[i] = xr[tid + i * 128]; s1 += x[i]; s2 += x[i] * x[i]; }
#pragma unroll
  for (int m = 1; m < 32; m <<= 1) { s1 += __shfl_xor(s1, m, 32); s2 += __shfl_xor(s2, m, 32); }
  int w = tid >> 5;
  if ((tid & 31) == 0) { red[w * 2] = s1; red[w * 2 + 1] = s2; }
  __syncthreads();
  s1 = red[0] + red[2] + red[4] + red[6];
  s2 = red[1] + red[3] + red[5] + red[7];
  float mean = s1 * (1.f / D_);
  float var  = s2 * (1.f / D_) - mean * mean;
  float rstd = rsqrtf(var + 1e-5f);
#pragma unroll
  for (int i = 0; i < 4; ++i) {
    int d = tid + i * 128;
    float y = (x[i] - mean) * rstd * g[d] + bb[d];
    if (LEAKY) y = (y >= 0.f) ? y : 0.2f * y;
    if (o32) o32[(size_t)r * D_ + d] = y;
    o16[(size_t)r * ld16 + d] = f2bf(y);
  }
}

// ---------------------------------------------------------------------------
// Timestep embedding MLP: pe[timesteps] -> silu MLP -> h rows t=0 (32 rows).
// ---------------------------------------------------------------------------
__global__ __launch_bounds__(128)
void ts_kernel(const int* __restrict__ tsteps, const float* __restrict__ Wt1,
               const float* __restrict__ bt1, const float* __restrict__ Wt2,
               const float* __restrict__ bt2, float* __restrict__ hf32) {
  __shared__ float pe_r[512];
  __shared__ float t1[512];
  const int b = blockIdx.x, tid = threadIdx.x;
  float pos = (float)tsteps[b];
  const float c0 = -logf(10000.f) / 512.f;
  for (int d = tid; d < 512; d += 128) {
    int j = d >> 1;
    float ang = pos * __expf((float)(2 * j) * c0);
    pe_r[d] = (d & 1) ? cosf(ang) : sinf(ang);
  }
  __syncthreads();
  for (int n = tid; n < 512; n += 128) {
    float a = bt1[n];
    for (int k = 0; k < 512; ++k) a += pe_r[k] * Wt1[k * 512 + n];
    t1[n] = a / (1.f + __expf(-a));  // silu
  }
  __syncthreads();
  for (int n = tid; n < 512; n += 128) {
    float a = bt2[n];
    for (int k = 0; k < 512; ++k) a += t1[k] * Wt2[k * 512 + n];
    hf32[(size_t)b * D_ + n] = a;
  }
}

// h[r] += pe[t], t = r/B; also produce bf16 copy of h.
__global__ __launch_bounds__(128)
void add_pe_kernel(float* __restrict__ hf32, unsigned short* __restrict__ hb16) {
  const int r = blockIdx.x, tid = threadIdx.x;
  float pos = (float)(r >> 5);
  const float c0 = -logf(10000.f) / 512.f;
  for (int d = tid; d < 512; d += 128) {
    int j = d >> 1;
    float ang = pos * __expf((float)(2 * j) * c0);
    float pe = (d & 1) ? cosf(ang) : sinf(ang);
    float v = hf32[(size_t)r * D_ + d] + pe;
    hf32[(size_t)r * D_ + d] = v;
    hb16[(size_t)r * D_ + d] = f2bf(v);
  }
}

// (B,S,F) fp32 -> (s*B+b, Fp) bf16 with zero pad.
__global__ __launch_bounds__(128)
void pack_tokens(const float* __restrict__ X, unsigned short* __restrict__ Y,
                 int F, int Fp) {
  const int row = blockIdx.x;           // s*B + b
  const int s = row >> 5, b = row & 31;
  const float* src = X + ((size_t)b * S_ + s) * F;
  unsigned short* dst = Y + (size_t)row * Fp;
  for (int f = threadIdx.x; f < Fp; f += 128)
    dst[f] = f2bf(f < F ? src[f] : 0.f);
}

// fp32 (K,N) -> bf16 (Kp,N) with zero-padded extra rows.
__global__ void cvt_pad_rows(const float* __restrict__ X, unsigned short* __restrict__ Y,
                             int K, int Kp, int N) {
  int i = blockIdx.x * 256 + threadIdx.x;
  if (i >= Kp * N) return;
  int k = i / N;
  Y[i] = f2bf(k < K ? X[i] : 0.f);
}

__global__ void cvt_kernel(const float* __restrict__ X, unsigned short* __restrict__ Y, int n) {
  int i = blockIdx.x * 256 + threadIdx.x;
  if (i < n) Y[i] = f2bf(X[i]);
}

// Output heads: pose (64->24) and expr (448->100), fp32.
__global__ __launch_bounds__(128)
void head_kernel(const float* __restrict__ hf32, const float* __restrict__ Wp,
                 const float* __restrict__ bp, const float* __restrict__ We,
                 const float* __restrict__ be, float* __restrict__ out) {
  __shared__ float sh[512];
  const int token = blockIdx.x, tid = threadIdx.x;
  const int b = token >> 9, s = token & 511;
  const float* src = hf32 + ((size_t)(s + 1) * B_ + b) * D_;
  for (int d = tid; d < 512; d += 128) sh[d] = src[d];
  __syncthreads();
  int c = tid;
  if (c < 24) {
    float a = bp[c];
    for (int k = 0; k < 64; ++k) a += sh[k] * Wp[k * 24 + c];
    out[(size_t)token * 124 + c] = a;
  } else if (c < 124) {
    int cc = c - 24;
    float a = be[cc];
    for (int k = 0; k < 448; ++k) a += sh[64 + k] * We[k * 100 + cc];
    out[(size_t)token * 124 + c] = a;
  }
}

// ---------------------------------------------------------------------------
extern "C" void kernel_launch(void* const* d_in, const int* in_sizes, int n_in,
                              void* d_out, int out_size, void* d_ws, size_t ws_size,
                              hipStream_t stream) {
  const float* x          = (const float*)d_in[0];
  const int*   tsteps     = (const int*)  d_in[1];
  const float* sparse     = (const float*)d_in[2];
  const float* W_lmk      = (const float*)d_in[3];
  const float* b_lmk      = (const float*)d_in[4];
  const float* g_lmk      = (const float*)d_in[5];
  const float* be_lmk     = (const float*)d_in[6];
  const float* W_in       = (const float*)d_in[7];
  const float* b_in       = (const float*)d_in[8];
  const float* g_in       = (const float*)d_in[9];
  const float* be_in      = (const float*)d_in[10];
  const float* W_merge    = (const float*)d_in[11];
  const float* b_merge    = (const float*)d_in[12];
  const float* Wt1        = (const float*)d_in[13];
  const float* bt1        = (const float*)d_in[14];
  const float* Wt2        = (const float*)d_in[15];
  const float* bt2        = (const float*)d_in[16];
  const float* in_proj_w  = (const float*)d_in[17];
  const float* in_proj_b  = (const float*)d_in[18];
  const float* out_proj_w = (const float*)d_in[19];
  const float* out_proj_b = (const float*)d_in[20];
  const float* ln1_g      = (const float*)d_in[21];
  const float* ln1_b      = (const float*)d_in[22];
  const float* ln2_g      = (const float*)d_in[23];
  const float* ln2_b      = (const float*)d_in[24];
  const float* W_ff1      = (const float*)d_in[25];
  const float* b_ff1      = (const float*)d_in[26];
  const float* W_ff2      = (const float*)d_in[27];
  const float* b_ff2      = (const float*)d_in[28];
  const float* W_pose     = (const float*)d_in[29];
  const float* b_pose     = (const float*)d_in[30];
  const float* W_expr     = (const float*)d_in[31];
  const float* b_expr     = (const float*)d_in[32];
  (void)in_sizes; (void)n_in; (void)out_size; (void)ws_size;

  char* ws = (char*)d_ws;
  size_t off = 0;
  auto alloc = [&](size_t bytes) -> void* {
    void* p = ws + off;
    off += (bytes + 255) & ~(size_t)255;
    return p;
  };

  float*          hf32   = (float*)alloc((size_t)MT_ * D_ * 4);
  float*          tmpA   = (float*)alloc((size_t)MT_ * D_ * 4);
  unsigned short* hb16   = (unsigned short*)alloc((size_t)MT_ * D_ * 2);
  unsigned short* qkv    = (unsigned short*)alloc((size_t)MT_ * 3 * D_ * 2);
  unsigned short* o16    = (unsigned short*)alloc((size_t)MT_ * D_ * 2);
  unsigned short* f116   = (unsigned short*)alloc((size_t)MT_ * FF_ * 2);
  unsigned short* Am     = (unsigned short*)alloc((size_t)MTOK_ * 1024 * 2);
  unsigned short* Asp    = (unsigned short*)alloc((size_t)MTOK_ * 224 * 2);
  unsigned short* Ax     = (unsigned short*)alloc((size_t)MTOK_ * 128 * 2);
  unsigned short* Wlmk16 = (unsigned short*)alloc((size_t)224 * 512 * 2);
  unsigned short* Win16  = (unsigned short*)alloc((size_t)128 * 512 * 2);
  unsigned short* Wmrg16 = (unsigned short*)alloc((size_t)1024 * 512 * 2);
  unsigned short* Wqkv16 = (unsigned short*)alloc((size_t)L_ * 512 * 1536 * 2);
  unsigned short* Wout16 = (unsigned short*)alloc((size_t)L_ * 512 * 512 * 2);
  unsigned short* Wff116 = (unsigned short*)alloc((size_t)L_ * 512 * 2048 * 2);
  unsigned short* Wff216 = (unsigned short*)alloc((size_t)L_ * 2048 * 512 * 2);

  const dim3 blk128(128);
  auto gemm_grid = [](int M, int N) { return dim3((N + 63) / 64, (M + 63) / 64); };
  auto cgrid = [](int n) { return dim3((n + 255) / 256); };

  // ---- weight conversion / padding to bf16 ----
  cvt_pad_rows<<<cgrid(224 * 512), 256, 0, stream>>>(W_lmk, Wlmk16, 204, 224, 512);
  cvt_pad_rows<<<cgrid(128 * 512), 256, 0, stream>>>(W_in, Win16, 124, 128, 512);
  cvt_kernel<<<cgrid(1024 * 512), 256, 0, stream>>>(W_merge, Wmrg16, 1024 * 512);
  cvt_kernel<<<cgrid(L_ * 512 * 1536), 256, 0, stream>>>(in_proj_w, Wqkv16, L_ * 512 * 1536);
  cvt_kernel<<<cgrid(L_ * 512 * 512), 256, 0, stream>>>(out_proj_w, Wout16, L_ * 512 * 512);
  cvt_kernel<<<cgrid(L_ * 512 * 2048), 256, 0, stream>>>(W_ff1, Wff116, L_ * 512 * 2048);
  cvt_kernel<<<cgrid(L_ * 2048 * 512), 256, 0, stream>>>(W_ff2, Wff216, L_ * 2048 * 512);

  // ---- input packing (transpose (B,S,.) -> (S*B,.) with K padding) ----
  pack_tokens<<<MTOK_, blk128, 0, stream>>>(sparse, Asp, 204, 224);
  pack_tokens<<<MTOK_, blk128, 0, stream>>>(x, Ax, 124, 128);

  // ---- input projections + LN + leaky, concat into Am ----
  gemm_bf16_k<0, false, true, false><<<gemm_grid(MTOK_, 512), blk128, 0, stream>>>(
      Asp, Wlmk16, b_lmk, nullptr, tmpA, nullptr, MTOK_, 512, 224);
  ln_kernel<true><<<MTOK_, blk128, 0, stream>>>(tmpA, g_lmk, be_lmk, nullptr, Am, 1024);
  gemm_bf16_k<0, false, true, false><<<gemm_grid(MTOK_, 512), blk128, 0, stream>>>(
      Ax, Win16, b_in, nullptr, tmpA, nullptr, MTOK_, 512, 128);
  ln_kernel<true><<<MTOK_, blk128, 0, stream>>>(tmpA, g_in, be_in, nullptr, Am + 512, 1024);

  // ---- merge projection -> h rows 32.. ; ts rows 0..31 ; + positional enc ----
  gemm_bf16_k<0, false, true, false><<<gemm_grid(MTOK_, 512), blk128, 0, stream>>>(
      Am, Wmrg16, b_merge, nullptr, hf32 + (size_t)B_ * D_, nullptr, MTOK_, 512, 1024);
  ts_kernel<<<B_, blk128, 0, stream>>>(tsteps, Wt1, bt1, Wt2, bt2, hf32);
  add_pe_kernel<<<MT_, blk128, 0, stream>>>(hf32, hb16);

  // ---- transformer layers ----
  for (int l = 0; l < L_; ++l) {
    gemm_bf16_k<0, false, false, true><<<gemm_grid(MT_, 1536), blk128, 0, stream>>>(
        hb16, Wqkv16 + (size_t)l * 512 * 1536, in_proj_b + (size_t)l * 1536,
        nullptr, nullptr, qkv, MT_, 1536, 512);
    attn_kernel<<<dim3(B_ * H_, (T_ + 63) / 64), blk128, 0, stream>>>(qkv, o16);
    gemm_bf16_k<0, true, true, false><<<gemm_grid(MT_, 512), blk128, 0, stream>>>(
        o16, Wout16 + (size_t)l * 512 * 512, out_proj_b + (size_t)l * 512,
        hf32, tmpA, nullptr, MT_, 512, 512);
    ln_kernel<false><<<MT_, blk128, 0, stream>>>(
        tmpA, ln1_g + (size_t)l * 512, ln1_b + (size_t)l * 512, hf32, hb16, 512);
    gemm_bf16_k<1, false, false, true><<<gemm_grid(MT_, 2048), blk128, 0, stream>>>(
        hb16, Wff116 + (size_t)l * 512 * 2048, b_ff1 + (size_t)l * 2048,
        nullptr, nullptr, f116, MT_, 2048, 512);
    gemm_bf16_k<0, true, true, false><<<gemm_grid(MT_, 512), blk128, 0, stream>>>(
        f116, Wff216 + (size_t)l * 2048 * 512, b_ff2 + (size_t)l * 512,
        hf32, tmpA, nullptr, MT_, 512, 2048);
    ln_kernel<false><<<MT_, blk128, 0, stream>>>(
        tmpA, ln2_g + (size_t)l * 512, ln2_b + (size_t)l * 512, hf32, hb16, 512);
  }

  // ---- output heads ----
  head_kernel<<<MTOK_, blk128, 0, stream>>>(hf32, W_pose, b_pose, W_expr, b_expr,
                                            (float*)d_out);
}
